// MultiHeadCrossAttention_38457137169076
// MI455X (gfx1250) — matrix operand order repaired and checked by reading, not compiled, not run on hardware
//
#include <hip/hip_runtime.h>
#include <hip/hip_bf16.h>
#include <cfloat>

// ---------------------------------------------------------------------------
// MultiHeadCrossAttention for MI455X (gfx1250): bf16 WMMA everywhere, fp32
// accumulate + fp32 online softmax. B=2, Tq=Tk=2048, E=1024, H=16, D=64.
// ---------------------------------------------------------------------------

typedef __bf16 bf16_t;
typedef __bf16 v16bf __attribute__((ext_vector_type(16)));
typedef float  v8f   __attribute__((ext_vector_type(8)));

#define LOG2E 1.4426950408889634f

static __device__ __forceinline__ v8f wmma_bf16(v16bf a, v16bf b, v8f c) {
  // D = A(16x32) * B(32x16) + C, f32 accumulate
  return __builtin_amdgcn_wmma_f32_16x16x32_bf16(false, a, false, b,
                                                 (short)0, c, false, false);
}

union FragU { uint4 q[2]; v16bf v; };

// A-fragment: 16x32 bf16 tile of row-major matrix (ld in elements).
// lane l: row = row0 + (l&15); 8 bf16 at k0+(l>>4)*8 and 8 at +16.
static __device__ __forceinline__ v16bf load_afrag(const bf16_t* base, int ld,
                                                   int row0, int k0) {
  const int lane = threadIdx.x & 31;
  const bf16_t* p = base + (size_t)(row0 + (lane & 15)) * ld + k0 + ((lane >> 4) * 8);
  FragU f;
  f.q[0] = *(const uint4*)(p);
  f.q[1] = *(const uint4*)(p + 16);
  return f.v;
}

// B-fragment: 32x16 operand, loaded from the TRANSPOSED matrix Bt[N][K]
// (row-major, ld in elements). lane l: column n = col0+(l&15);
// 8 bf16 at k0+(l>>4)*8 and 8 at +16.
static __device__ __forceinline__ v16bf load_bfrag(const bf16_t* bt, int ld,
                                                   int col0, int k0) {
  const int lane = threadIdx.x & 31;
  const bf16_t* p = bt + (size_t)(col0 + (lane & 15)) * ld + k0 + ((lane >> 4) * 8);
  FragU f;
  f.q[0] = *(const uint4*)(p);
  f.q[1] = *(const uint4*)(p + 16);
  return f.v;
}

// ---------------------------------------------------------------------------
// Elementwise cast f32 -> bf16
__global__ __launch_bounds__(256)
void cast_f32_bf16(const float* __restrict__ in, bf16_t* __restrict__ out, int n) {
  int i = blockIdx.x * blockDim.x + threadIdx.x;
  if (i < n) out[i] = (bf16_t)in[i];
}

// Transpose-cast weight w[K=1024][Ncols] f32 -> wt[Ncols][1024] bf16
__global__ __launch_bounds__(256)
void cast_wt_1024(const float* __restrict__ w, bf16_t* __restrict__ wt,
                  int n, int Ncols) {
  int i = blockIdx.x * blockDim.x + threadIdx.x;
  if (i < n) {
    int col = i >> 10;       // output row = weight column
    int k   = i & 1023;      // K index (K == 1024 for all weights here)
    wt[i] = (bf16_t)w[(size_t)k * Ncols + col];
  }
}

// ---------------------------------------------------------------------------
// GEMM: C[M,N] = A[M,K](bf16,row-major) * Bt[N,K](bf16, pre-transposed B).
// One wave computes a 64x64 C tile (16 accumulators). blockDim = 128 (4 waves).
// mode 0: C0 = bf16(acc*scale), row-major [M,N]
// mode 1: KV split: col<1024 -> K_flat bf16 [4096,1024];
//                   col>=1024 -> Vt bf16 [(b*1024 + h*64+d), t] (ld 2048)
// mode 2: C0 = f32 acc, row-major [M,N]
__global__ __launch_bounds__(128)
void gemm_wmma_kernel(const bf16_t* __restrict__ A, const bf16_t* __restrict__ Bt,
                      int M, int N, int K,
                      void* __restrict__ C0, void* __restrict__ C1,
                      float scale, int mode) {
  const int wave = (blockIdx.x * blockDim.x + threadIdx.x) >> 5;
  const int mtiles = M >> 6;
  const int m0 = (wave % mtiles) * 64;
  const int n0 = (wave / mtiles) * 64;

  v8f acc[4][4] = {};
  for (int kk = 0; kk < K; kk += 32) {
    v16bf a[4], b[4];
#pragma unroll
    for (int i = 0; i < 4; i++) a[i] = load_afrag(A, K, m0 + 16 * i, kk);
#pragma unroll
    for (int j = 0; j < 4; j++) b[j] = load_bfrag(Bt, K, n0 + 16 * j, kk);
#pragma unroll
    for (int i = 0; i < 4; i++)
#pragma unroll
      for (int j = 0; j < 4; j++)
        acc[i][j] = wmma_bf16(a[i], b[j], acc[i][j]);
  }

  const int lane = threadIdx.x & 31;
  const int nl = lane & 15;          // C-layout: column = lane
  const int mh = (lane >> 4) * 8;    // row offset for upper lane half
#pragma unroll
  for (int i = 0; i < 4; i++)
#pragma unroll
    for (int j = 0; j < 4; j++)
#pragma unroll
      for (int r = 0; r < 8; r++) {
        int row = m0 + 16 * i + mh + r;
        int col = n0 + 16 * j + nl;
        float v = acc[i][j][r] * scale;
        if (mode == 0) {
          ((bf16_t*)C0)[(size_t)row * N + col] = (bf16_t)v;
        } else if (mode == 1) {
          if (col < 1024) {
            ((bf16_t*)C0)[(size_t)row * 1024 + col] = (bf16_t)v;   // K features
          } else {
            int hd = col - 1024;            // h*64 + d
            int t  = row & 2047;
            int bb = row >> 11;
            ((bf16_t*)C1)[((size_t)(bb * 1024 + hd)) * 2048 + t] = (bf16_t)v; // V^T
          }
        } else {
          ((float*)C0)[(size_t)row * N + col] = v;
        }
      }
}

// ---------------------------------------------------------------------------
// Flash attention: one wave per (b, h, 16-query tile). Computes S^T = K * Q^T
// so the C layout puts a fixed query per lane -> lane-local P fragment build
// and a single xor-16 shuffle per softmax reduction.
// Qf: [4096,1024] bf16, pre-scaled by SCALE*log2e. Kf: [4096,1024] bf16.
// Vt: [(b*1024 + h*64 + d), 2048] bf16. Of: [4096,1024] bf16 (head-interleaved).
__global__ __launch_bounds__(128)
void attn_kernel(const bf16_t* __restrict__ Qf, const bf16_t* __restrict__ Kf,
                 const bf16_t* __restrict__ Vt, const float* __restrict__ wrel,
                 bf16_t* __restrict__ Of) {
  const int wave = (blockIdx.x * blockDim.x + threadIdx.x) >> 5;
  const int lane = threadIdx.x & 31;
  const int qt = wave & 127;     // Tq/16 = 128 tiles
  const int bh = wave >> 7;      // b*16 + h
  const int h = bh & 15;
  const int b = bh >> 4;
  const int q0 = qt * 16;
  const int rowq0 = b * 2048;
  const float wr = wrel[h] * LOG2E;   // bias slope in log2 units

  const int nl = lane & 15;
  const int mh = (lane >> 4) * 8;
  const float qpos = (float)(q0 + nl);

  const bf16_t* Qb  = Qf + (size_t)rowq0 * 1024;
  const bf16_t* Kb  = Kf + (size_t)rowq0 * 1024;
  const bf16_t* Vbh = Vt + (size_t)(b * 1024 + h * 64) * 2048;

  // Q^T B-fragments, reused for every key chunk (K-dim = D = 64 -> 2 frags)
  v16bf qb0 = load_bfrag(Qb, 1024, q0, h * 64);
  v16bf qb1 = load_bfrag(Qb, 1024, q0, h * 64 + 32);

  v8f o0 = {}, o1 = {}, o2 = {}, o3 = {};   // O tile: 16q x 64d, f32
  float mrun = -FLT_MAX, lrun = 0.f;

  for (int kc = 0; kc < 2048; kc += 32) {
    if (kc + 32 < 2048) {  // hint next K/V chunks toward the caches
      __builtin_prefetch(Kb + (size_t)(kc + 32) * 1024 + h * 64, 0, 0);
      __builtin_prefetch(Vbh + (size_t)nl * 2048 + kc + 32, 0, 0);
    }
    // S^T tiles: (keys kc..kc+15) x (16 queries) and (keys kc+16..kc+31)
    v16bf a0 = load_afrag(Kb, 1024, kc,      h * 64);
    v16bf a1 = load_afrag(Kb, 1024, kc,      h * 64 + 32);
    v16bf a2 = load_afrag(Kb, 1024, kc + 16, h * 64);
    v16bf a3 = load_afrag(Kb, 1024, kc + 16, h * 64 + 32);
    v8f z = {};
    v8f st0 = wmma_bf16(a0, qb0, z);
    st0 = wmma_bf16(a1, qb1, st0);
    v8f st1 = wmma_bf16(a2, qb0, z);
    st1 = wmma_bf16(a3, qb1, st1);

    // bias + online softmax (all in log2 units; Q pre-scaled by SCALE*log2e)
    float s[16];
#pragma unroll
    for (int r = 0; r < 8; r++) {
      s[r]     = st0[r] + (qpos - (float)(kc + mh + r)) * wr;
      s[8 + r] = st1[r] + (qpos - (float)(kc + 16 + mh + r)) * wr;
    }
    float mx = s[0];
#pragma unroll
    for (int j = 1; j < 16; j++) mx = fmaxf(mx, s[j]);
    mx = fmaxf(mx, __shfl_xor(mx, 16, 32));     // reduce across lane halves
    float mnew = fmaxf(mrun, mx);
    float alpha = exp2f(mrun - mnew);
    float rs = 0.f;
    v16bf p;                                    // P A-fragment, lane-local
#pragma unroll
    for (int j = 0; j < 16; j++) {
      float e = exp2f(s[j] - mnew);
      rs += e;
      p[j] = (bf16_t)e;
    }
    rs += __shfl_xor(rs, 16, 32);
    lrun = lrun * alpha + rs;
    mrun = mnew;

    // rescale O rows: O row m needs alpha of query m (held by lane m)
    float am[8];
#pragma unroll
    for (int r = 0; r < 8; r++) am[r] = __shfl(alpha, mh + r, 32);
#pragma unroll
    for (int r = 0; r < 8; r++) {
      o0[r] *= am[r]; o1[r] *= am[r]; o2[r] *= am[r]; o3[r] *= am[r];
    }

    // O += P(16x32) * V(32x64): V^T rows are D, contiguous over keys
    v16bf v0 = load_bfrag(Vbh, 2048, 0,  kc);
    v16bf v1 = load_bfrag(Vbh, 2048, 16, kc);
    v16bf v2 = load_bfrag(Vbh, 2048, 32, kc);
    v16bf v3 = load_bfrag(Vbh, 2048, 48, kc);
    o0 = wmma_bf16(p, v0, o0);
    o1 = wmma_bf16(p, v1, o1);
    o2 = wmma_bf16(p, v2, o2);
    o3 = wmma_bf16(p, v3, o3);
  }

  // normalize by row sums and store bf16 head-interleaved output
  float inv = 1.0f / lrun;
  float bm[8];
#pragma unroll
  for (int r = 0; r < 8; r++) bm[r] = __shfl(inv, mh + r, 32);
  bf16_t* orow = Of + (size_t)(rowq0 + q0) * 1024 + h * 64;
#pragma unroll
  for (int r = 0; r < 8; r++) {
    size_t ro = (size_t)(mh + r) * 1024;
    orow[ro +  0 + nl] = (bf16_t)(o0[r] * bm[r]);
    orow[ro + 16 + nl] = (bf16_t)(o1[r] * bm[r]);
    orow[ro + 32 + nl] = (bf16_t)(o2[r] * bm[r]);
    orow[ro + 48 + nl] = (bf16_t)(o3[r] * bm[r]);
  }
}

// ---------------------------------------------------------------------------
extern "C" void kernel_launch(void* const* d_in, const int* in_sizes, int n_in,
                              void* d_out, int out_size, void* d_ws, size_t ws_size,
                              hipStream_t stream) {
  const float* x_q    = (const float*)d_in[0];
  const float* x_kv   = (const float*)d_in[1];
  const float* w_q    = (const float*)d_in[2];
  const float* w_kv   = (const float*)d_in[3];
  const float* w_proj = (const float*)d_in[4];
  const float* w_rel  = (const float*)d_in[5];

  char* ws = (char*)d_ws;
  const size_t MB = 1024 * 1024;
  bf16_t* Xq   = (bf16_t*)(ws + 0);         // 8 MB  [4096,1024]
  bf16_t* Xkv  = (bf16_t*)(ws + 8 * MB);    // 8 MB  [4096,1024]
  bf16_t* WqT  = (bf16_t*)(ws + 16 * MB);   // 2 MB  [1024,1024]
  bf16_t* WkvT = (bf16_t*)(ws + 18 * MB);   // 4 MB  [2048,1024]
  bf16_t* WpT  = (bf16_t*)(ws + 22 * MB);   // 2 MB  [1024,1024]
  bf16_t* Qf   = (bf16_t*)(ws + 24 * MB);   // 8 MB  [4096,1024] (pre-scaled)
  bf16_t* Kf   = (bf16_t*)(ws + 32 * MB);   // 8 MB  [4096,1024]
  bf16_t* Vt   = (bf16_t*)(ws + 40 * MB);   // 8 MB  [2048 rows, 2048]
  bf16_t* Of   = (bf16_t*)(ws + 48 * MB);   // 8 MB  [4096,1024]

  const int NE = 4096 * 1024;
  cast_f32_bf16<<<NE / 256, 256, 0, stream>>>(x_q,  Xq,  NE);
  cast_f32_bf16<<<NE / 256, 256, 0, stream>>>(x_kv, Xkv, NE);
  cast_wt_1024<<<(1024 * 1024) / 256, 256, 0, stream>>>(w_q,    WqT,  1024 * 1024, 1024);
  cast_wt_1024<<<(2048 * 1024) / 256, 256, 0, stream>>>(w_kv,   WkvT, 2048 * 1024, 2048);
  cast_wt_1024<<<(1024 * 1024) / 256, 256, 0, stream>>>(w_proj, WpT,  1024 * 1024, 1024);

  const float qscale = 0.125f * LOG2E;  // HEAD_DIM^-0.5 * log2(e), folded into Q

  // Q = Xq @ Wq (scaled), 1024 wave tiles -> 256 blocks of 4 waves
  gemm_wmma_kernel<<<256, 128, 0, stream>>>(Xq, WqT, 4096, 1024, 1024,
                                            (void*)Qf, nullptr, qscale, 0);
  // KV = Xkv @ Wkv, split epilogue into K (row-major) and V (transposed)
  gemm_wmma_kernel<<<512, 128, 0, stream>>>(Xkv, WkvT, 4096, 2048, 1024,
                                            (void*)Kf, (void*)Vt, 1.0f, 1);
  // attention: 4096 waves (B*H*Tq/16) -> 1024 blocks of 4 waves
  attn_kernel<<<1024, 128, 0, stream>>>(Qf, Kf, Vt, w_rel, Of);
  // out = O @ Wproj, f32 to d_out
  gemm_wmma_kernel<<<256, 128, 0, stream>>>(Of, WpT, 4096, 1024, 1024,
                                            d_out, nullptr, 1.0f, 2);
}
